// Decoder_3289944948800
// MI455X (gfx1250) — compile-verified, matrix operand order checked by reading, not run on hardware
//
#include <hip/hip_runtime.h>

// Problem constants (from reference)
#define B_SZ  256     // batch
#define H_SZ  512     // LSTM hidden
#define G_SZ  2048    // 4*H gates
#define T_SZ  512     // trg length (511 LSTM steps)
#define HID   1024    // fc1 out
#define OUTD  32000   // fc2 out
#define NBLK  128     // persistent LSTM grid (must all be resident)
#define NTHR  256     // 8 waves of 32

typedef __attribute__((ext_vector_type(2)))  float   v2f;
typedef __attribute__((ext_vector_type(4)))  float   v4f;
typedef __attribute__((ext_vector_type(8)))  float   v8f;
typedef __attribute__((ext_vector_type(4)))  __bf16  v4bf;
typedef __attribute__((ext_vector_type(8)))  __bf16  v8bf;
typedef __attribute__((ext_vector_type(16))) __bf16  v16bf;

__device__ __forceinline__ float sigf(float x)  { return 1.0f / (1.0f + __expf(-x)); }
__device__ __forceinline__ float tanhf_(float x){ return 1.0f - 2.0f / (__expf(2.0f * x) + 1.0f); }

__device__ __forceinline__ v16bf cat8(v8bf lo, v8bf hi) {
  return __builtin_shufflevector(lo, hi, 0,1,2,3,4,5,6,7,8,9,10,11,12,13,14,15);
}

// ---------------------------------------------------------------------------
// Software grid barrier (persistent-kernel style; all NBLK blocks resident).
// ---------------------------------------------------------------------------
__device__ __forceinline__ void grid_sync(unsigned* cnt, unsigned* gen) {
  __threadfence();      // make this block's writes device-visible
  __syncthreads();
  if (threadIdx.x == 0) {
    unsigned g = __hip_atomic_load(gen, __ATOMIC_RELAXED, __HIP_MEMORY_SCOPE_AGENT);
    unsigned a = __hip_atomic_fetch_add(cnt, 1u, __ATOMIC_ACQ_REL, __HIP_MEMORY_SCOPE_AGENT) + 1u;
    if (a == (unsigned)NBLK) {
      __hip_atomic_store(cnt, 0u, __ATOMIC_RELAXED, __HIP_MEMORY_SCOPE_AGENT);
      __hip_atomic_store(gen, g + 1u, __ATOMIC_RELEASE, __HIP_MEMORY_SCOPE_AGENT);
    } else {
      while (__hip_atomic_load(gen, __ATOMIC_ACQUIRE, __HIP_MEMORY_SCOPE_AGENT) == g)
        __builtin_amdgcn_s_sleep(2);
    }
  }
  __syncthreads();
  __threadfence();      // acquire side: invalidate stale lines before reads
}

// ---------------------------------------------------------------------------
// Kernel 1: prep — zero sync + state, bias = b_ih + b_hh, w_hh -> bf16.
// ---------------------------------------------------------------------------
__global__ void prep_kernel(const float* __restrict__ b_ih, const float* __restrict__ b_hh,
                            const float* __restrict__ w_hh,
                            float* __restrict__ bias, __bf16* __restrict__ whhB,
                            float* __restrict__ hF, float* __restrict__ cF,
                            __bf16* __restrict__ hB, unsigned* __restrict__ sync) {
  int gid = blockIdx.x * blockDim.x + threadIdx.x;
  int stride = gridDim.x * blockDim.x;
  if (gid < 64) sync[gid] = 0u;           // d_ws is poisoned 0xAA: barrier counter must start 0
  for (int i = gid; i < G_SZ; i += stride) bias[i] = b_ih[i] + b_hh[i];
  for (int i = gid; i < G_SZ * H_SZ; i += stride) whhB[i] = (__bf16)w_hh[i];
  for (int i = gid; i < B_SZ * H_SZ; i += stride) {
    hF[i] = 0.0f; cF[i] = 0.0f; hB[i] = (__bf16)0.0f;
  }
}

// ---------------------------------------------------------------------------
// Kernel 2: persistent LSTM. Block jt owns gate columns [jt*16, jt*16+16).
// Per step: bf16 WMMA GEMM (K=512 in 16 x v_wmma_f32_16x16x32_bf16 per tile),
// grid barrier, fused elementwise LSTM cell update, grid barrier.
// Weight strip (16 rows x 512 K, 16KB) staged in LDS once via async-to-LDS.
// ---------------------------------------------------------------------------
__global__ void __launch_bounds__(NTHR)
lstm_kernel(const int* __restrict__ trg, const float* __restrict__ w_ih,
            const float* __restrict__ bias, const __bf16* __restrict__ whhB,
            float* __restrict__ hF, float* __restrict__ cF, __bf16* __restrict__ hB,
            float* __restrict__ gates, unsigned* __restrict__ sync) {
  __shared__ __bf16 ldsW[16 * H_SZ];   // 16 KB weight strip for this block's N-tile
  const int jt = blockIdx.x;           // N-tile (gate column group), 0..127

  // --- stage weight strip into LDS with gfx1250 async-to-LDS DMA ---
  {
    unsigned base = (unsigned)(unsigned long long)(void*)&ldsW[0];
    const __bf16* src = whhB + (size_t)jt * 16 * H_SZ;
    for (int i = threadIdx.x; i < 2048; i += NTHR) {     // 2048 x 16B chunks
      unsigned dst = base + (unsigned)(i * 16);
      unsigned long long ga = (unsigned long long)(const void*)(src + i * 8);
      asm volatile("global_load_async_to_lds_b128 %0, %1, off"
                   :: "v"(dst), "v"(ga) : "memory");
    }
    asm volatile("s_wait_asynccnt 0x0" ::: "memory");
  }
  __syncthreads();

  const int wv     = threadIdx.x >> 5;   // wave 0..7, owns M-tiles {2wv, 2wv+1}
  const int lane   = threadIdx.x & 31;
  const int lane15 = lane & 15;
  const int kb     = (lane >> 4) * 8;    // K sub-phase per half-wave
  const __bf16* brow = &ldsW[lane15 * H_SZ];         // column lane15 of W^T strip
  const int m0  = (wv * 2) * 16 + lane15;
  const int m1  = (wv * 2 + 1) * 16 + lane15;
  const int col = jt * 16 + lane15;

  unsigned* cnt = sync + 0;
  unsigned* gen = sync + 1;

  // elementwise-phase mapping: 32768 threads x 4 consecutive units
  const int gid = blockIdx.x * NTHR + threadIdx.x;
  const int eb  = gid >> 7;
  const int eu  = (gid & 127) * 4;

  #pragma unroll 1
  for (int t = 0; t < T_SZ - 1; ++t) {
    // -------- phase 1: gates[:, strip] = h_bf16 @ w_hh_strip^T --------
    v8f acc0 = {0,0,0,0,0,0,0,0};
    v8f acc1 = {0,0,0,0,0,0,0,0};
    const __bf16* a0p = hB + (size_t)m0 * H_SZ;
    const __bf16* a1p = hB + (size_t)m1 * H_SZ;
    #pragma unroll
    for (int kc = 0; kc < 16; ++kc) {
      const int k0 = kc * 32 + kb;
      v16bf bfrag = cat8(*(const v8bf*)(brow + k0), *(const v8bf*)(brow + k0 + 16));
      v16bf a0    = cat8(*(const v8bf*)(a0p  + k0), *(const v8bf*)(a0p  + k0 + 16));
      v16bf a1    = cat8(*(const v8bf*)(a1p  + k0), *(const v8bf*)(a1p  + k0 + 16));
      acc0 = __builtin_amdgcn_wmma_f32_16x16x32_bf16(false, a0, false, bfrag, (short)0, acc0, false, false);
      acc1 = __builtin_amdgcn_wmma_f32_16x16x32_bf16(false, a1, false, bfrag, (short)0, acc1, false, false);
    }
    #pragma unroll
    for (int r = 0; r < 8; ++r) {
      gates[(size_t)(wv * 32      + kb + r) * G_SZ + col] = acc0[r];
      gates[(size_t)(wv * 32 + 16 + kb + r) * G_SZ + col] = acc1[r];
    }
    grid_sync(cnt, gen);

    // -------- phase 2: LSTM cell update (i,f,g,o -> c,h) --------
    {
      const float xt = (float)trg[eb * T_SZ + t];
      const float* gb = gates + (size_t)eb * G_SZ;
      v4f gi = *(const v4f*)(gb +         eu);
      v4f gf = *(const v4f*)(gb +  512 +  eu);
      v4f gg = *(const v4f*)(gb + 1024 +  eu);
      v4f go = *(const v4f*)(gb + 1536 +  eu);
      v4f wi = *(const v4f*)(w_ih +        eu);
      v4f wf = *(const v4f*)(w_ih +  512 + eu);
      v4f wg = *(const v4f*)(w_ih + 1024 + eu);
      v4f wo = *(const v4f*)(w_ih + 1536 + eu);
      v4f bi = *(const v4f*)(bias +        eu);
      v4f bq = *(const v4f*)(bias +  512 + eu);
      v4f bg = *(const v4f*)(bias + 1024 + eu);
      v4f bo = *(const v4f*)(bias + 1536 + eu);
      v4f cc = *(v4f*)(cF + (size_t)eb * H_SZ + eu);
      v4f hh;
      v4bf hb4;
      #pragma unroll
      for (int e = 0; e < 4; ++e) {
        float iv = sigf  (gi[e] + xt * wi[e] + bi[e]);
        float fv = sigf  (gf[e] + xt * wf[e] + bq[e]);
        float gv = tanhf_(gg[e] + xt * wg[e] + bg[e]);
        float ov = sigf  (go[e] + xt * wo[e] + bo[e]);
        float cv = fv * cc[e] + iv * gv;
        float hv = ov * tanhf_(cv);
        cc[e] = cv; hh[e] = hv; hb4[e] = (__bf16)hv;
      }
      *(v4f*) (cF + (size_t)eb * H_SZ + eu) = cc;
      *(v4f*) (hF + (size_t)eb * H_SZ + eu) = hh;
      *(v4bf*)(hB + (size_t)eb * H_SZ + eu) = hb4;
    }
    grid_sync(cnt, gen);
  }
}

// ---------------------------------------------------------------------------
// Kernel 3: z = relu(h_last @ fc1_w^T + fc1_b), fp32 WMMA (16x16x4_f32).
// 1024 output tiles; 128 blocks x 8 waves, one tile per wave.
// ---------------------------------------------------------------------------
__global__ void __launch_bounds__(NTHR)
fc1_kernel(const float* __restrict__ hF, const float* __restrict__ w1,
           const float* __restrict__ b1, float* __restrict__ zF) {
  const int wv = threadIdx.x >> 5, lane = threadIdx.x & 31;
  const int gw = blockIdx.x * (NTHR / 32) + wv;   // 0..1023
  const int mt = gw & 15, nt = gw >> 4;
  const int lane15 = lane & 15;
  const int klo = (lane >> 4) * 2;
  const float* arow = hF + (size_t)(mt * 16 + lane15) * H_SZ;
  const float* brow = w1 + (size_t)(nt * 16 + lane15) * H_SZ;
  v8f acc = {0,0,0,0,0,0,0,0};
  #pragma unroll 4
  for (int kc = 0; kc < H_SZ / 4; ++kc) {
    const int ks = kc * 4 + klo;
    v2f a = *(const v2f*)(arow + ks);
    v2f b = *(const v2f*)(brow + ks);
    acc = __builtin_amdgcn_wmma_f32_16x16x4_f32(false, a, false, b, (short)0, acc, false, false);
  }
  const int colN = nt * 16 + lane15;
  const int rowoff = (lane >> 4) * 8;
  const float bb = b1[colN];
  #pragma unroll
  for (int r = 0; r < 8; ++r) {
    float v = acc[r] + bb;
    zF[(size_t)(mt * 16 + rowoff + r) * HID + colN] = v > 0.0f ? v : 0.0f;
  }
}

// ---------------------------------------------------------------------------
// Kernel 4: out = z @ fc2_w^T + fc2_b. HBM-bound on the 128MB fp32 weight
// read -> keep fp32, one pass, fp32 WMMA. 2000 blocks, each owns 16 columns;
// 8 waves x 2 M-tiles (B-fragment reused across both WMMAs).
// ---------------------------------------------------------------------------
__global__ void __launch_bounds__(NTHR)
fc2_kernel(const float* __restrict__ zF, const float* __restrict__ w2,
           const float* __restrict__ b2, float* __restrict__ out) {
  const int wv = threadIdx.x >> 5, lane = threadIdx.x & 31;
  const int nt = blockIdx.x;                 // 0..1999
  const int lane15 = lane & 15;
  const int klo = (lane >> 4) * 2;
  const int colN = nt * 16 + lane15;
  const float* brow = w2 + (size_t)colN * HID;
  const float* a0p  = zF + (size_t)(wv * 16        + lane15) * HID;
  const float* a1p  = zF + (size_t)((wv + 8) * 16  + lane15) * HID;
  v8f acc0 = {0,0,0,0,0,0,0,0};
  v8f acc1 = {0,0,0,0,0,0,0,0};
  #pragma unroll 4
  for (int kc = 0; kc < HID / 4; ++kc) {
    const int ks = kc * 4 + klo;
    if ((kc & 31) == 0) __builtin_prefetch(brow + ks + 256, 0, 1);  // global_prefetch
    v2f b  = *(const v2f*)(brow + ks);
    v2f a0 = *(const v2f*)(a0p + ks);
    v2f a1 = *(const v2f*)(a1p + ks);
    acc0 = __builtin_amdgcn_wmma_f32_16x16x4_f32(false, a0, false, b, (short)0, acc0, false, false);
    acc1 = __builtin_amdgcn_wmma_f32_16x16x4_f32(false, a1, false, b, (short)0, acc1, false, false);
  }
  const int rowoff = (lane >> 4) * 8;
  const float bb = b2[colN];
  #pragma unroll
  for (int r = 0; r < 8; ++r) {
    out[(size_t)(wv * 16       + rowoff + r) * OUTD + colN] = acc0[r] + bb;
    out[(size_t)((wv + 8) * 16 + rowoff + r) * OUTD + colN] = acc1[r] + bb;
  }
}

// ---------------------------------------------------------------------------
extern "C" void kernel_launch(void* const* d_in, const int* in_sizes, int n_in,
                              void* d_out, int out_size, void* d_ws, size_t ws_size,
                              hipStream_t stream) {
  // Input order: x, hidden, trg, w_ih, w_hh, b_ih, b_hh, fc1_w, fc1_b, fc2_w, fc2_b
  const int*   trg   = (const int*)  d_in[2];
  const float* w_ih  = (const float*)d_in[3];
  const float* w_hh  = (const float*)d_in[4];
  const float* b_ih  = (const float*)d_in[5];
  const float* b_hh  = (const float*)d_in[6];
  const float* fc1_w = (const float*)d_in[7];
  const float* fc1_b = (const float*)d_in[8];
  const float* fc2_w = (const float*)d_in[9];
  const float* fc2_b = (const float*)d_in[10];
  float* out = (float*)d_out;

  // Workspace carve-up (~6.3 MB total), 256B aligned
  char* ws = (char*)d_ws;
  size_t off = 0;
  auto take = [&](size_t bytes) -> char* {
    off = (off + 255) & ~(size_t)255;
    char* p = ws + off;
    off += bytes;
    return p;
  };
  unsigned* sync  = (unsigned*)take(256);
  float*    bias  = (float*)   take((size_t)G_SZ * 4);
  __bf16*   whhB  = (__bf16*)  take((size_t)G_SZ * H_SZ * 2);
  float*    hF    = (float*)   take((size_t)B_SZ * H_SZ * 4);
  float*    cF    = (float*)   take((size_t)B_SZ * H_SZ * 4);
  __bf16*   hB    = (__bf16*)  take((size_t)B_SZ * H_SZ * 2);
  float*    gates = (float*)   take((size_t)B_SZ * G_SZ * 4);
  float*    zF    = (float*)   take((size_t)B_SZ * HID * 4);
  (void)ws_size; (void)in_sizes; (void)n_in; (void)out_size;

  prep_kernel<<<512, 256, 0, stream>>>(b_ih, b_hh, w_hh, bias, whhB, hF, cF, hB, sync);
  lstm_kernel<<<NBLK, NTHR, 0, stream>>>(trg, w_ih, bias, whhB, hF, cF, hB, gates, sync);
  fc1_kernel<<<(16 * (HID / 16)) / (NTHR / 32), NTHR, 0, stream>>>(hF, fc1_w, fc1_b, zF);
  fc2_kernel<<<OUTD / 16, NTHR, 0, stream>>>(zF, fc2_w, fc2_b, out);
}
